// GATLayer_28389733826665
// MI455X (gfx1250) — matrix operand order repaired and checked by reading, not compiled, not run on hardware
//
#include <hip/hip_runtime.h>
#include <hip/hip_bf16.h>

#define NROW  8192
#define INF_  256
#define OUTF  128
#define ALPHA 0.02f

typedef __attribute__((ext_vector_type(16))) _Float16 v16h;
typedef __attribute__((ext_vector_type(8)))  float    v8f;
typedef __attribute__((ext_vector_type(4)))  unsigned v4u;
typedef __attribute__((ext_vector_type(2)))  unsigned v2u;
typedef __attribute__((ext_vector_type(4)))  float    v4f;
typedef __attribute__((ext_vector_type(4)))  int      v4i;

union FragH { v16h h; v4u u[2]; };

static __device__ __forceinline__ unsigned pack2(float lo, float hi) {
    union { _Float16 h[2]; unsigned u; } p;
    p.h[0] = (_Float16)lo; p.h[1] = (_Float16)hi;
    return p.u;
}

// ---------------------------------------------------------------------------
// k0: swizzle W1 (f32) into WMMA B-fragment layout, packed f16.
// Fragment f = kt*8 + w (K-tile kt, 16-col slice w). Per lane (n=lane&15,
// hi=lane>>4), dword d packs rows {k0+16*hi+2d, +1} of column w*16+n.
// ---------------------------------------------------------------------------
__global__ void k0_w1_swizzle(const float* __restrict__ W1,
                              unsigned* __restrict__ W1B) {
    int g = blockIdx.x * 256 + threadIdx.x;   // 0..2047 (64 frags * 32 lanes)
    int f = g >> 5, lane = g & 31;
    int kt = f >> 3, w = f & 7;
    int n = lane & 15, hi = lane >> 4;
    int col = w * 16 + n;
    int base = (f * 32 + lane) * 8;
    for (int d = 0; d < 8; ++d) {
        int r0 = kt * 32 + hi * 16 + 2 * d;
        W1B[base + d] = pack2(W1[r0 * OUTF + col], W1[(r0 + 1) * OUTF + col]);
    }
}

// ---------------------------------------------------------------------------
// k1: Wh = x @ W1 + b1.  Block = 8 waves; wave wv owns 16x16 tile (rows
// blockIdx*16, cols wv*16). Stores Wh f32 and Wh f16 pre-swizzled into the
// B-fragment layout k2 consumes (LDS bounce for the cross-lane shuffle).
// ---------------------------------------------------------------------------
__global__ void __launch_bounds__(256)
k1_gemm_wh(const float* __restrict__ x, const unsigned* __restrict__ W1B,
           const float* __restrict__ b1, float* __restrict__ Wh,
           unsigned* __restrict__ WhB) {
    __shared__ _Float16 sT[8][16][17];
    int tid = threadIdx.x;
    int wv = tid >> 5, lane = tid & 31;
    int m = lane & 15, hi = lane >> 4;
    int i0 = blockIdx.x * 16;
    int n0 = wv * 16;

    v8f C = {};
    for (int k = 0; k < INF_; k += 32) {
        FragH A, B;
        // A: elements 0..15 = K k+16*hi+0..15
        const float* xp = x + (size_t)(i0 + m) * INF_ + k + hi * 16;
        v4f x0 = *(const v4f*)(xp);
        v4f x1 = *(const v4f*)(xp + 4);
        v4f x2 = *(const v4f*)(xp + 8);
        v4f x3 = *(const v4f*)(xp + 12);
        for (int t = 0; t < 4; ++t) {
            A.h[t]      = (_Float16)x0[t];
            A.h[4 + t]  = (_Float16)x1[t];
            A.h[8 + t]  = (_Float16)x2[t];
            A.h[12 + t] = (_Float16)x3[t];
        }
        const unsigned* bp = W1B + (size_t)((((k >> 5) * 8 + wv) * 32 + lane) * 8);
        B.u[0] = *(const v4u*)(bp);
        B.u[1] = *(const v4u*)(bp + 4);
        C = __builtin_amdgcn_wmma_f32_16x16x32_f16(false, A.h, false, B.h,
                                                   (short)0, C, false, false);
    }
    float bias = b1[n0 + m];
    for (int v = 0; v < 8; ++v) C[v] += bias;

    // f32 Wh for the attention-vector GEMVs
    for (int v = 0; v < 8; ++v)
        Wh[(size_t)(i0 + v + 8 * hi) * OUTF + n0 + m] = C[v];

    // bounce C tile through LDS to rearrange into B-fragment order
    for (int v = 0; v < 8; ++v)
        sT[wv][v + 8 * hi][m] = (_Float16)C[v];
    __syncthreads();
    if (lane < 16) {
        int jt = blockIdx.x >> 1;                    // j-tile this row tile feeds
        int lane2 = ((blockIdx.x & 1) << 4) | lane;  // which half of the 32 K-rows
        unsigned d[8];
        for (int r = 0; r < 8; ++r)
            d[r] = pack2((float)sT[wv][2 * r][lane], (float)sT[wv][2 * r + 1][lane]);
        unsigned* op = WhB + (size_t)(((jt * 8 + wv) * 32 + lane2) * 8);
        *(v4u*)(op)     = (v4u){d[0], d[1], d[2], d[3]};
        *(v4u*)(op + 4) = (v4u){d[4], d[5], d[6], d[7]};
    }
}

// ---------------------------------------------------------------------------
// k1b: f_src[j] = Wh[j,:]@a[:128], f_dst[j] = Wh[j,:]@a[128:]. Wave per row.
// ---------------------------------------------------------------------------
__global__ void __launch_bounds__(256)
k1b_attn_vec(const float* __restrict__ Wh, const float* __restrict__ a,
             float* __restrict__ fsrc, float* __restrict__ fdst) {
    int wv = threadIdx.x >> 5, lane = threadIdx.x & 31;
    int j = blockIdx.x * 8 + wv;
    v4f w4 = *(const v4f*)(Wh + (size_t)j * OUTF + lane * 4);
    v4f a0 = *(const v4f*)(a + lane * 4);
    v4f a1 = *(const v4f*)(a + OUTF + lane * 4);
    float s0 = w4[0]*a0[0] + w4[1]*a0[1] + w4[2]*a0[2] + w4[3]*a0[3];
    float s1 = w4[0]*a1[0] + w4[1]*a1[1] + w4[2]*a1[2] + w4[3]*a1[3];
    for (int off = 16; off; off >>= 1) {
        s0 += __shfl_xor(s0, off, 32);
        s1 += __shfl_xor(s1, off, 32);
    }
    if (lane == 0) { fsrc[j] = s0; fdst[j] = s1; }
}

// k1c: global max of f_dst (for the exact softmax upper bound)
__global__ void k1c_max(const float* __restrict__ fdst, float* __restrict__ Mg) {
    __shared__ float red[256];
    float m = -1e30f;
    for (int t = threadIdx.x; t < NROW; t += 256) m = fmaxf(m, fdst[t]);
    red[threadIdx.x] = m;
    __syncthreads();
    for (int s = 128; s; s >>= 1) {
        if (threadIdx.x < s)
            red[threadIdx.x] = fmaxf(red[threadIdx.x], red[threadIdx.x + s]);
        __syncthreads();
    }
    if (threadIdx.x == 0) Mg[0] = red[0];
}

// ---------------------------------------------------------------------------
// k2: fused masked softmax + (attention @ Wh) + ELU.
// Block = 16 output rows, 256 threads = 8 waves (wave wv owns cols wv*16..+15).
// Stream j in tiles of 64: each thread builds 4 masked exp() values (one int4
// adj load, one 8B packed-f16 LDS store), then each wave issues two
// v_wmma_f32_16x16x32_f16: C += P[16x64] @ Wh[64x16] per barrier pair.
// sP row stride = 72 halves (144B = 9*16B): b128-aligned, conflict-free
// (36*m mod 64 distinct for m=0..15) at all four chunk offsets.
// ---------------------------------------------------------------------------
__global__ void __launch_bounds__(256)
k2_gat_attention(const int* __restrict__ adj, const unsigned* __restrict__ WhB,
                 const float* __restrict__ fsrc, const float* __restrict__ fdst,
                 const float* __restrict__ Mg, const float* __restrict__ batt,
                 float* __restrict__ out) {
    __shared__ _Float16 sP[16][72];
    __shared__ float sDen[16];
    int tid = threadIdx.x;
    int wv = tid >> 5, lane = tid & 31;
    int m = lane & 15, hi = lane >> 4;
    int n0 = wv * 16;
    int i0 = blockIdx.x * 16;
    int pr = tid >> 4;            // P row this thread produces
    int pc = (tid & 15) * 4;      // P cols (4 per thread)

    float s  = fsrc[i0 + pr] + batt[0];
    float mu = s + Mg[0];
    mu = mu > 0.f ? mu : ALPHA * mu;   // exact row upper bound (lrelu monotone)
    float dsum = 0.f;
    if (tid < 16) sDen[tid] = 0.f;

    v8f C = {};
    const size_t arow = (size_t)(i0 + pr) * NROW;
    for (int jt2 = 0; jt2 < NROW / 64; ++jt2) {
        int j0 = jt2 * 64;
        v4i a4 = *(const v4i*)(adj + arow + j0 + pc);
        v4f f4 = *(const v4f*)(fdst + j0 + pc);
        if (jt2 + 1 < NROW / 64)
            __builtin_prefetch(adj + arow + j0 + 64 + pc, 0, 1);
        float p[4];
        float d01 = 0.f, d23 = 0.f;
        for (int t = 0; t < 4; ++t) {
            float e = s + f4[t];
            e = e > 0.f ? e : ALPHA * e;              // LeakyReLU
            p[t] = a4[t] > 0 ? __expf(e - mu) : 0.f;  // masked softmax numer
        }
        d01 = p[0] + p[1]; d23 = p[2] + p[3];
        dsum += d01 + d23;

        __syncthreads();               // previous iteration's A-frag reads done
        *(v2u*)&sP[pr][pc] = (v2u){pack2(p[0], p[1]), pack2(p[2], p[3])};
        __syncthreads();

        FragH A0, A1, B0, B1;
        A0.u[0] = *(const v4u*)&sP[m][hi * 16];        // K j0+  0..31
        A0.u[1] = *(const v4u*)&sP[m][hi * 16 + 8];
        A1.u[0] = *(const v4u*)&sP[m][32 + hi * 16];   // K j0+ 32..63
        A1.u[1] = *(const v4u*)&sP[m][40 + hi * 16];
        const unsigned* bp0 = WhB + (size_t)((((2 * jt2) * 8 + wv) * 32 + lane) * 8);
        const unsigned* bp1 = WhB + (size_t)((((2 * jt2 + 1) * 8 + wv) * 32 + lane) * 8);
        B0.u[0] = *(const v4u*)(bp0);
        B0.u[1] = *(const v4u*)(bp0 + 4);
        B1.u[0] = *(const v4u*)(bp1);
        B1.u[1] = *(const v4u*)(bp1 + 4);
        C = __builtin_amdgcn_wmma_f32_16x16x32_f16(false, A0.h, false, B0.h,
                                                   (short)0, C, false, false);
        C = __builtin_amdgcn_wmma_f32_16x16x32_f16(false, A1.h, false, B1.h,
                                                   (short)0, C, false, false);
    }

    atomicAdd(&sDen[pr], dsum);        // 16 partials per row -> denominator
    __syncthreads();
    for (int v = 0; v < 8; ++v) {
        int row = v + 8 * hi;          // C layout: VGPR v = M rows v / v+8
        float h = C[v] / sDen[row];
        out[(size_t)(i0 + row) * OUTF + n0 + m] = h > 0.f ? h : __expf(h) - 1.f;
    }
}

// ---------------------------------------------------------------------------
extern "C" void kernel_launch(void* const* d_in, const int* in_sizes, int n_in,
                              void* d_out, int out_size, void* d_ws, size_t ws_size,
                              hipStream_t stream) {
    const float* x    = (const float*)d_in[0];
    const int*   adj  = (const int*)d_in[1];
    const float* W1   = (const float*)d_in[2];
    const float* b1   = (const float*)d_in[3];
    const float* a    = (const float*)d_in[4];
    const float* batt = (const float*)d_in[5];
    float* out = (float*)d_out;

    // workspace: Wh f32 (4MB) | WhB f16-packed (2MB) | W1B (64KB) | vecs
    float*    Wh   = (float*)d_ws;
    unsigned* WhB  = (unsigned*)(Wh + (size_t)NROW * OUTF);
    unsigned* W1B  = WhB + (size_t)NROW * OUTF / 2;
    float*    fsrc = (float*)(W1B + (size_t)INF_ * OUTF / 2);
    float*    fdst = fsrc + NROW;
    float*    Mg   = fdst + NROW;

    k0_w1_swizzle<<<8, 256, 0, stream>>>(W1, W1B);
    k1_gemm_wh<<<NROW / 16, 256, 0, stream>>>(x, W1B, b1, Wh, WhB);
    k1b_attn_vec<<<NROW / 8, 256, 0, stream>>>(Wh, a, fsrc, fdst);
    k1c_max<<<1, 256, 0, stream>>>(fdst, Mg);
    k2_gat_attention<<<NROW / 16, 256, 0, stream>>>(adj, WhB, fsrc, fdst, Mg,
                                                    batt, out);
}
